// GlideAttention_39788577030586
// MI455X (gfx1250) — compile-verified, matrix operand order checked by reading, not run on hardware
//
#include <hip/hip_runtime.h>

#define S_LEN 2048
#define HID   1024
#define HD    64
#define NH    16
#define NKV   4
#define DKV   256   // NKV*HD

typedef __attribute__((ext_vector_type(8)))  __bf16 bf16x8;
typedef __attribute__((ext_vector_type(16))) __bf16 bf16x16;
typedef __attribute__((ext_vector_type(8)))  float  f32x8;
typedef __attribute__((ext_vector_type(4)))  unsigned int uint32x4;
typedef __attribute__((ext_vector_type(8)))  int    int32x8;
typedef __attribute__((ext_vector_type(4)))  int    int32x4;

__device__ __forceinline__ unsigned short f2bf(float f) {
  unsigned int u = __float_as_uint(f);
  unsigned int r = u + 0x7FFFu + ((u >> 16) & 1u);   // round-to-nearest-even
  return (unsigned short)(r >> 16);
}

// Load a 16-bit WMMA operand fragment (A 16x32 rows-M, or B 32x16 cols-N) from a
// row-major buffer; fragment "index" selects a row of leading dim `ld` (elements),
// K contiguous. Layout per CDNA5 ISA 7.12.2.
__device__ __forceinline__ bf16x16 load_frag(const unsigned short* p, int ld) {
  int lane = threadIdx.x & 31;
  int r  = lane & 15;
  int kb = (lane < 16) ? 0 : 8;
  const bf16x8 lo = *(const bf16x8*)(p + r * ld + kb);
  const bf16x8 hi = *(const bf16x8*)(p + r * ld + kb + 16);
  bf16x16 f;
#pragma unroll
  for (int i = 0; i < 8; ++i) { f[i] = lo[i]; f[i + 8] = hi[i]; }
  return f;
}

__device__ __forceinline__ f32x8 wmma_bf16(bf16x16 a, bf16x16 b, f32x8 c) {
  return __builtin_amdgcn_wmma_f32_16x16x32_bf16(false, a, false, b, (short)0, c,
                                                 false, false);
}

// ---- TDM D# group1 pack (ISA §8.4): data_size=2B, 2D tile, dim0-stride only ----
__device__ __forceinline__ int32x8 tdm_g1(unsigned td0, unsigned td1,
                                          unsigned tile0, unsigned tile1,
                                          unsigned long long stride0) {
  int32x8 g = {0, 0, 0, 0, 0, 0, 0, 0};
  g[0] = (int)(1u << 16);                                        // data_size=2B
  g[1] = (int)((td0 & 0xFFFFu) << 16);                           // tensor_dim0[15:0]
  g[2] = (int)(((td0 >> 16) & 0xFFFFu) | ((td1 & 0xFFFFu) << 16));
  g[3] = (int)(((td1 >> 16) & 0xFFFFu) | ((tile0 & 0xFFFFu) << 16));
  g[4] = (int)(tile1 & 0xFFFFu);                                 // tile_dim1; dim2=0
  g[5] = (int)(unsigned)(stride0 & 0xFFFFFFFFu);                 // dim0_stride lo
  g[6] = (int)(unsigned)((stride0 >> 32) & 0xFFFFu);             // dim0_stride hi
  g[7] = 0;
  return g;
}

__device__ __forceinline__ uint32x4 tdm_g0(unsigned lds_off,
                                           unsigned long long gaddr) {
  uint32x4 g;
  g[0] = 1u;                                                     // count=1
  g[1] = lds_off;                                                // lds_addr
  g[2] = (unsigned)(gaddr & 0xFFFFFFFFu);                        // global_addr lo
  g[3] = (unsigned)(((gaddr >> 32) & 0x01FFFFFFu) | 0x80000000u);// hi + type=2
  return g;
}

// ---------------- stage 1: fp32 -> bf16 converts ----------------
__global__ void cvt_kernel(const float* __restrict__ in,
                           unsigned short* __restrict__ out, int n) {
  int i = blockIdx.x * 256 + threadIdx.x;
  if (i < n) out[i] = f2bf(in[i]);
}

// out[n*K + k] = in[k*N + n]   (transpose + convert)
__global__ void cvt_t_kernel(const float* __restrict__ in,
                             unsigned short* __restrict__ out, int K, int N) {
  int o = blockIdx.x * 256 + threadIdx.x;
  if (o < K * N) {
    int k = o % K;
    int n = o / K;
    out[o] = f2bf(in[k * N + n]);
  }
}

// ---------------- stage 2: Q/K projection + bias + RoPE ----------------
__global__ __launch_bounds__(256) void proj_rope_kernel(
    const unsigned short* __restrict__ Xb, const unsigned short* __restrict__ Wt,
    const float* __restrict__ bias, const float* __restrict__ cosp,
    const float* __restrict__ sinp, unsigned short* __restrict__ out, int Nout) {
  int wave = (blockIdx.x * 256 + threadIdx.x) >> 5;
  int tiles_n = Nout >> 6;
  int mt = wave / tiles_n, nt = wave % tiles_n;
  int s0 = mt << 4, n0 = nt << 6;
  int lane = threadIdx.x & 31;

  f32x8 acc[4];
#pragma unroll
  for (int j = 0; j < 4; ++j) acc[j] = (f32x8){0, 0, 0, 0, 0, 0, 0, 0};

  for (int k0 = 0; k0 < HID; k0 += 32) {
    __builtin_prefetch((const void*)(Xb + s0 * HID + k0 + 64), 0, 1);
    bf16x16 a = load_frag(Xb + s0 * HID + k0, HID);
#pragma unroll
    for (int j = 0; j < 4; ++j) {
      bf16x16 b = load_frag(Wt + (n0 + j * 16) * HID + k0, HID);
      acc[j] = wmma_bf16(a, b, acc[j]);
    }
  }

  int ncol = lane & 15;
  int mofs = (lane >> 4) << 3;  // C layout: M = v + mofs
  int head = n0 >> 6;
#pragma unroll
  for (int j = 0; j < 2; ++j) {
    int dlo = j * 16 + ncol;
    int dhi = dlo + 32;
    float bl = bias[n0 + dlo];
    float bh = bias[n0 + dhi];
#pragma unroll
    for (int v = 0; v < 8; ++v) {
      int s = s0 + v + mofs;
      float c  = cosp[s * HD + dlo];
      float sn = sinp[s * HD + dlo];
      float qlo = acc[j][v] + bl;
      float qhi = acc[j + 2][v] + bh;
      out[(head * S_LEN + s) * HD + dlo] = f2bf(qlo * c - qhi * sn);
      out[(head * S_LEN + s) * HD + dhi] = f2bf(qhi * c + qlo * sn);
    }
  }
}

// ---------------- stage 3: V projection, stored transposed [kv*64+d][s] ----------------
__global__ __launch_bounds__(256) void proj_vT_kernel(
    const unsigned short* __restrict__ Xb, const unsigned short* __restrict__ Wt,
    const float* __restrict__ bias, unsigned short* __restrict__ outT, int Nout) {
  int wave = (blockIdx.x * 256 + threadIdx.x) >> 5;
  int tiles_n = Nout >> 6;
  int mt = wave / tiles_n, nt = wave % tiles_n;
  int s0 = mt << 4, n0 = nt << 6;
  int lane = threadIdx.x & 31;

  f32x8 acc[4];
#pragma unroll
  for (int j = 0; j < 4; ++j) acc[j] = (f32x8){0, 0, 0, 0, 0, 0, 0, 0};

  for (int k0 = 0; k0 < HID; k0 += 32) {
    bf16x16 a = load_frag(Xb + s0 * HID + k0, HID);
#pragma unroll
    for (int j = 0; j < 4; ++j) {
      bf16x16 b = load_frag(Wt + (n0 + j * 16) * HID + k0, HID);
      acc[j] = wmma_bf16(a, b, acc[j]);
    }
  }

  int ncol = lane & 15;
  int mofs = (lane >> 4) << 3;
#pragma unroll
  for (int j = 0; j < 4; ++j) {
    int n = n0 + j * 16 + ncol;
    float bv = bias[n];
#pragma unroll
    for (int v = 0; v < 8; ++v) {
      int s = s0 + v + mofs;
      outT[n * S_LEN + s] = f2bf(acc[j][v] + bv);
    }
  }
}

// ---------------- stage 4: flash attention, TDM-staged K/V in LDS ----------------
// 8 waves/block all share one KV head; wave w handles q-tile q0blk + 16*w.
__global__ __launch_bounds__(256) void attn_kernel(
    const unsigned short* __restrict__ Q, const unsigned short* __restrict__ K,
    const unsigned short* __restrict__ Vt, unsigned short* __restrict__ attnb) {
  __shared__ __align__(16) unsigned short ldsP[8][16 * 32];  // per-wave P tiles
  __shared__ __align__(16) unsigned short kbuf[32 * HD];     // K chunk  [32 s][64 d]
  __shared__ __align__(16) unsigned short vbuf[HD * 32];     // Vt chunk [64 d][32 s]

  int wid = threadIdx.x >> 5;
  int lane = threadIdx.x & 31;
  int h = blockIdx.x >> 4;                 // 16 blocks per head
  int q0blk = (blockIdx.x & 15) << 7;      // 128 q rows per block
  int q0 = q0blk + (wid << 4);
  int hk = h >> 2;                         // GROUPS = 4

  bf16x16 aq0 = load_frag(Q + (h * S_LEN + q0) * HD, HD);
  bf16x16 aq1 = load_frag(Q + (h * S_LEN + q0) * HD + 32, HD);

  f32x8 o[4];
  float m[8], l[8];
#pragma unroll
  for (int j = 0; j < 4; ++j) o[j] = (f32x8){0, 0, 0, 0, 0, 0, 0, 0};
#pragma unroll
  for (int v = 0; v < 8; ++v) { m[v] = -1e30f; l[v] = 0.0f; }

  int ncol = lane & 15;
  int mofs = (lane >> 4) << 3;
  unsigned short* Pl = &ldsP[wid][0];
  int kend = q0 + 12;           // this wave's block-causal KV bound
  int kend_blk = q0blk + 124;   // block-max bound

  for (int c0 = 0; c0 < kend_blk; c0 += 32) {
    // ---- wave 0: DMA the shared K / Vt chunks into LDS via the TDM ----
    if (wid == 0) {
#if __has_builtin(__builtin_amdgcn_tensor_load_to_lds)
      unsigned kl = (unsigned)(unsigned long long)(uintptr_t)&kbuf[0];
      unsigned vl = (unsigned)(unsigned long long)(uintptr_t)&vbuf[0];
      unsigned long long kg =
          (unsigned long long)(uintptr_t)(K + (size_t)(hk * S_LEN + c0) * HD);
      unsigned long long vg =
          (unsigned long long)(uintptr_t)(Vt + (size_t)(hk * HD) * S_LEN + c0);
      int32x4 zero4 = {0, 0, 0, 0};
      int32x8 zero8 = {0, 0, 0, 0, 0, 0, 0, 0};
      // K chunk: 32 rows x 64 cols, row stride 64 elements
      __builtin_amdgcn_tensor_load_to_lds(
          tdm_g0(kl, kg), tdm_g1(HD, 65536u, HD, 32u, (unsigned long long)HD),
          zero4, zero4, zero8, 0);
      // Vt chunk: 64 rows (d) x 32 cols (s), row stride S elements
      __builtin_amdgcn_tensor_load_to_lds(
          tdm_g0(vl, vg), tdm_g1(S_LEN, 65536u, 32u, HD, (unsigned long long)S_LEN),
          zero4, zero4, zero8, 0);
      __builtin_amdgcn_s_wait_tensorcnt(0);
#else
      // fallback: plain cooperative copy
      for (int i = lane; i < 32 * HD / 8; i += 32) {
        *(bf16x8*)&kbuf[i * 8] =
            *(const bf16x8*)(K + (size_t)(hk * S_LEN + c0) * HD + i * 8);
      }
      for (int r = 0; r < HD; ++r) {
        if (lane < 4)
          *(bf16x8*)&vbuf[r * 32 + lane * 8] =
              *(const bf16x8*)(Vt + (size_t)(hk * HD + r) * S_LEN + c0 + lane * 8);
      }
#endif
    }
    __syncthreads();

    if (c0 < kend) {
      f32x8 s[2];
#pragma unroll
      for (int j = 0; j < 2; ++j) {
        f32x8 z = (f32x8){0, 0, 0, 0, 0, 0, 0, 0};
        bf16x16 bk0 = load_frag(&kbuf[(j * 16) * HD], HD);
        bf16x16 bk1 = load_frag(&kbuf[(j * 16) * HD + 32], HD);
        z = wmma_bf16(aq0, bk0, z);
        z = wmma_bf16(aq1, bk1, z);
        s[j] = z;
      }
      // scale + block mask: (q>>2) > (k>>2) || (q<4 && k<4)
#pragma unroll
      for (int j = 0; j < 2; ++j) {
        int kc = c0 + j * 16 + ncol;
#pragma unroll
        for (int v = 0; v < 8; ++v) {
          int q = q0 + v + mofs;
          bool ok = ((q >> 2) > (kc >> 2)) || ((q < 4) && (kc < 4));
          s[j][v] = ok ? s[j][v] * 0.125f : -1e30f;
        }
      }
      // online softmax
      float p0[8], p1[8];
#pragma unroll
      for (int v = 0; v < 8; ++v) {
        float mx = fmaxf(s[0][v], s[1][v]);
#pragma unroll
        for (int d = 1; d < 16; d <<= 1) mx = fmaxf(mx, __shfl_xor(mx, d, 32));
        float mn = fmaxf(m[v], mx);
        float sc = __expf(m[v] - mn);
        m[v] = mn;
        float e0 = __expf(s[0][v] - mn);
        float e1 = __expf(s[1][v] - mn);
        float rs = e0 + e1;
#pragma unroll
        for (int d = 1; d < 16; d <<= 1) rs += __shfl_xor(rs, d, 32);
        l[v] = l[v] * sc + rs;
        o[0][v] *= sc; o[1][v] *= sc; o[2][v] *= sc; o[3][v] *= sc;
        p0[v] = e0; p1[v] = e1;
      }
      // C-layout -> A-layout through per-wave LDS tile
#pragma unroll
      for (int v = 0; v < 8; ++v) {
        int M = v + mofs;
        Pl[M * 32 + ncol]      = f2bf(p0[v]);
        Pl[M * 32 + 16 + ncol] = f2bf(p1[v]);
      }
      asm volatile("s_wait_dscnt 0" ::: "memory");
      bf16x16 ap = load_frag(Pl, 32);
#pragma unroll
      for (int j2 = 0; j2 < 4; ++j2) {
        bf16x16 bv = load_frag(&vbuf[(j2 * 16) * 32], 32);
        o[j2] = wmma_bf16(ap, bv, o[j2]);
      }
      asm volatile("s_wait_dscnt 0" ::: "memory");
    }
    __syncthreads();
  }

  // finalize: divide by l, write bf16 attn output in [s][h*64+d] layout
#pragma unroll
  for (int v = 0; v < 8; ++v) {
    float inv = 1.0f / l[v];
    int s = q0 + v + mofs;
#pragma unroll
    for (int j = 0; j < 4; ++j) {
      attnb[s * HID + h * HD + j * 16 + ncol] = f2bf(o[j][v] * inv);
    }
  }
}

// ---------------- stage 5: output projection, fp32 out ----------------
__global__ __launch_bounds__(256) void out_gemm_kernel(
    const unsigned short* __restrict__ A, const unsigned short* __restrict__ Bt,
    float* __restrict__ out) {
  int wave = (blockIdx.x * 256 + threadIdx.x) >> 5;
  int mt = wave / (HID >> 6), nt = wave % (HID >> 6);
  int s0 = mt << 4, n0 = nt << 6;
  int lane = threadIdx.x & 31;

  f32x8 acc[4];
#pragma unroll
  for (int j = 0; j < 4; ++j) acc[j] = (f32x8){0, 0, 0, 0, 0, 0, 0, 0};

  for (int k0 = 0; k0 < HID; k0 += 32) {
    __builtin_prefetch((const void*)(A + s0 * HID + k0 + 64), 0, 1);
    bf16x16 a = load_frag(A + s0 * HID + k0, HID);
#pragma unroll
    for (int j = 0; j < 4; ++j) {
      bf16x16 b = load_frag(Bt + (n0 + j * 16) * HID + k0, HID);
      acc[j] = wmma_bf16(a, b, acc[j]);
    }
  }
  int ncol = lane & 15;
  int mofs = (lane >> 4) << 3;
#pragma unroll
  for (int j = 0; j < 4; ++j) {
#pragma unroll
    for (int v = 0; v < 8; ++v) {
      out[(s0 + v + mofs) * HID + n0 + j * 16 + ncol] = acc[j][v];
    }
  }
}

extern "C" void kernel_launch(void* const* d_in, const int* in_sizes, int n_in,
                              void* d_out, int out_size, void* d_ws, size_t ws_size,
                              hipStream_t stream) {
  const float* hs   = (const float*)d_in[0];
  const float* cosp = (const float*)d_in[1];
  const float* sinp = (const float*)d_in[2];
  const float* Wq   = (const float*)d_in[3];
  const float* bq   = (const float*)d_in[4];
  const float* Wk   = (const float*)d_in[5];
  const float* bk   = (const float*)d_in[6];
  const float* Wv   = (const float*)d_in[7];
  const float* bv   = (const float*)d_in[8];
  const float* Wo   = (const float*)d_in[9];
  float* out = (float*)d_out;

  char* ws = (char*)d_ws;
  unsigned short* Xb  = (unsigned short*)(ws);                               // 4 MB
  unsigned short* WqT = (unsigned short*)(ws + (4u << 20));                  // 2 MB
  unsigned short* WkT = (unsigned short*)(ws + (6u << 20));                  // 0.5 MB
  unsigned short* WvT = (unsigned short*)(ws + (6u << 20) + (512u << 10));   // 0.5 MB
  unsigned short* WoT = (unsigned short*)(ws + (7u << 20));                  // 2 MB
  unsigned short* Qb  = (unsigned short*)(ws + (9u << 20));                  // 4 MB
  unsigned short* Kb  = (unsigned short*)(ws + (13u << 20));                 // 1 MB
  unsigned short* Vt  = (unsigned short*)(ws + (14u << 20));                 // 1 MB
  unsigned short* Ab  = (unsigned short*)(ws + (15u << 20));                 // 4 MB

  cvt_kernel<<<(S_LEN * HID + 255) / 256, 256, 0, stream>>>(hs, Xb, S_LEN * HID);
  cvt_t_kernel<<<(HID * HID + 255) / 256, 256, 0, stream>>>(Wq, WqT, HID, HID);
  cvt_t_kernel<<<(HID * DKV + 255) / 256, 256, 0, stream>>>(Wk, WkT, HID, DKV);
  cvt_t_kernel<<<(HID * DKV + 255) / 256, 256, 0, stream>>>(Wv, WvT, HID, DKV);
  cvt_t_kernel<<<(HID * HID + 255) / 256, 256, 0, stream>>>(Wo, WoT, HID, HID);

  proj_rope_kernel<<<256, 256, 0, stream>>>(Xb, WqT, bq, cosp, sinp, Qb, HID);
  proj_rope_kernel<<<64, 256, 0, stream>>>(Xb, WkT, bk, cosp, sinp, Kb, DKV);
  proj_vT_kernel<<<64, 256, 0, stream>>>(Xb, WvT, bv, Vt, DKV);

  attn_kernel<<<256, 256, 0, stream>>>(Qb, Kb, Vt, Ab);
  out_gemm_kernel<<<256, 256, 0, stream>>>(Ab, WoT, out);
}
